// CRANModel_37641093382852
// MI455X (gfx1250) — matrix operand order repaired and checked by reading, not compiled, not run on hardware
//
#include <hip/hip_runtime.h>
#include <hip/hip_bf16.h>
#include <cstdint>
#include <cstddef>

// ---------------------------------------------------------------------------
// CDNA5 (gfx1250) CRAN model:
//   phase 0: prep  — transpose+convert weights f32 -> bf16 [N][K] layout
//   phase 1: recurrent — 1 workgroup (32 waves) runs all 64 steps with WMMA
//   phase 2: logits GEMM [2048x512]@[512x32000] bf16 WMMA, f32 accum + bias
// ---------------------------------------------------------------------------

typedef __bf16        v16bf __attribute__((ext_vector_type(16)));
typedef float         v8f   __attribute__((ext_vector_type(8)));
typedef unsigned int  u32x4 __attribute__((ext_vector_type(4)));

union BFrag { v16bf v; u32x4 q[2]; };

__device__ __forceinline__ unsigned short f2bf(float f) {
  unsigned int u = __float_as_uint(f);
  unsigned int r = u + 0x7FFFu + ((u >> 16) & 1u);   // round-to-nearest-even
  return (unsigned short)(r >> 16);
}
__device__ __forceinline__ float bf2f(unsigned short h) {
  return __uint_as_float(((unsigned int)h) << 16);
}

// A fragment 16x32 bf16, source row-major [M][K]:
// lanes 0-15 : v0-3 = K 0..7,  v4-7 = K 16..23   (rows m0..m0+15)
// lanes 16-31: v0-3 = K 8..15, v4-7 = K 24..31
__device__ __forceinline__ v16bf load_a16x32(const unsigned short* p, int m0,
                                             int k0, int stride, int lane) {
  const int row = m0 + (lane & 15);
  const int kb  = k0 + ((lane >> 4) << 3);
  const unsigned short* s = p + (size_t)row * stride + kb;
  BFrag f;
  f.q[0] = *(const u32x4*)(s);
  f.q[1] = *(const u32x4*)(s + 16);
  return f.v;
}

// B fragment 32x16 bf16, source stored as [N][K] row-major (i.e. B^T):
// lanes 0-15 : v0-3 = K 0..7,  v4-7 = K 16..23   (column n0+lane)
// lanes 16-31: v0-3 = K 8..15, v4-7 = K 24..31
__device__ __forceinline__ v16bf load_b32x16(const unsigned short* p, int n0,
                                             int k0, int stride, int lane) {
  const int col = n0 + (lane & 15);
  const int kb  = k0 + ((lane >> 4) << 3);
  const unsigned short* s = p + (size_t)col * stride + kb;
  BFrag f;
  f.q[0] = *(const u32x4*)(s);
  f.q[1] = *(const u32x4*)(s + 16);
  return f.v;
}

__device__ __forceinline__ v8f wmma_bf16(v16bf a, v16bf b, v8f c) {
  return __builtin_amdgcn_wmma_f32_16x16x32_bf16(false, a, false, b,
                                                 (short)0, c, false, false);
}

// ---------------------------------------------------------------------------
// prep: transpose f32 [R][C] -> bf16 [C][R]
// ---------------------------------------------------------------------------
__global__ __launch_bounds__(256) void transpose_f32_to_bf16(
    const float* __restrict__ src, unsigned short* __restrict__ dst,
    int R, int C) {
  __shared__ float tile[32][33];
  const int cx = blockIdx.x * 32 + threadIdx.x;   // src col
  const int ry = blockIdx.y * 32 + threadIdx.y;   // src row base
#pragma unroll
  for (int i = 0; i < 32; i += 8) {
    int r = ry + i;
    tile[threadIdx.y + i][threadIdx.x] =
        (r < R && cx < C) ? src[(size_t)r * C + cx] : 0.f;
  }
  __syncthreads();
  const int dx  = blockIdx.y * 32 + threadIdx.x;  // dst col (= src row)
  const int dyb = blockIdx.x * 32 + threadIdx.y;  // dst row (= src col)
#pragma unroll
  for (int i = 0; i < 32; i += 8) {
    int dr = dyb + i;
    if (dr < C && dx < R)
      dst[(size_t)dr * R + dx] = f2bf(tile[threadIdx.x][threadIdx.y + i]);
  }
}

__global__ __launch_bounds__(256) void cvt_f32_to_bf16(
    const float* __restrict__ src, unsigned short* __restrict__ dst, int n) {
  int i = blockIdx.x * blockDim.x + threadIdx.x;
  if (i < n) dst[i] = f2bf(src[i]);
}

// ---------------------------------------------------------------------------
// phase 1: recurrent core. One workgroup, 32 waves (wave32), 64 steps.
// dims: B=32, E=512, DK=256, DV=512, H=512, N=512, k=8, T=64
// K-chunk loops are kept rolled: full unroll at 8 waves/SIMD register budget
// caused the compiler to hoist all B fragments and spill them to scratch.
// ---------------------------------------------------------------------------
__global__ __launch_bounds__(1024) void cran_recurrent(
    const int* __restrict__ tokens, const float* __restrict__ emb,
    const unsigned short* __restrict__ WqT,   // [256][512] bf16
    const unsigned short* __restrict__ WhT,   // [512][1024] bf16
    const float* __restrict__ bh, const float* __restrict__ Wk,  // [512][256] f32
    const float* __restrict__ Wv,                                // [512][512] f32
    unsigned short* __restrict__ keys_bf,     // [512][256] bf16 (mutable cache)
    float* __restrict__ vals,                 // [512][512] f32  (mutable cache)
    unsigned short* __restrict__ hA)          // [2048][512] bf16 out
{
  __shared__ unsigned short cbuf[32 * 1024];  // concat(x, retrieved) bf16
  __shared__ unsigned short qbf[32 * 256];    // q bf16
  __shared__ float sbuf[32 * 512];            // scores, later reused for h (f32)
  __shared__ float wbuf[32 * 8];
  __shared__ int   ibuf[32 * 8];
  __shared__ float meanx[512];
  __shared__ float meanh[512];

  const int tid  = threadIdx.x;
  const int lane = tid & 31;
  const int wid  = tid >> 5;

  for (int t = 0; t < 64; ++t) {
    // --- 1. embedding gather: cbuf[r][0:512] = bf16(emb[tok]) -------------
    {
      const int r = wid;
      const int tok = tokens[t * 32 + r];
      const float* er = emb + (size_t)tok * 512;
#pragma unroll
      for (int i = 0; i < 16; ++i) {
        int e = lane + i * 32;
        cbuf[r * 1024 + e] = f2bf(er[e]);
      }
    }
    __syncthreads();

    // --- 2. q = x @ Wq  (M=32,N=256,K=512): 2x16 tiles, one per wave ------
    {
      const int mf = wid >> 4, nf = wid & 15;
      v8f acc = {};
#pragma clang loop unroll(disable)
      for (int kc = 0; kc < 16; ++kc) {
        v16bf a = load_a16x32(cbuf, mf * 16, kc * 32, 1024, lane);
        v16bf b = load_b32x16(WqT, nf * 16, kc * 32, 512, lane);
        acc = wmma_bf16(a, b, acc);
      }
      const int col = nf * 16 + (lane & 15);
      const int rb  = mf * 16 + ((lane >> 4) << 3);
#pragma unroll
      for (int i = 0; i < 8; ++i) qbf[(rb + i) * 256 + col] = f2bf(acc[i]);
    }
    __syncthreads();

    // --- 3. scores = (q @ keys^T) * 1/sqrt(256)  (M=32,N=512,K=256) -------
    {
#pragma clang loop unroll(disable)
      for (int rep = 0; rep < 2; ++rep) {
        const int tt = wid + rep * 32;
        const int mf = tt >> 5, nf = tt & 31;
        v8f acc = {};
#pragma clang loop unroll(disable)
        for (int kc = 0; kc < 8; ++kc) {
          v16bf a = load_a16x32(qbf, mf * 16, kc * 32, 256, lane);
          v16bf b = load_b32x16(keys_bf, nf * 16, kc * 32, 256, lane);
          acc = wmma_bf16(a, b, acc);
        }
        const int col = nf * 16 + (lane & 15);
        const int rb  = mf * 16 + ((lane >> 4) << 3);
#pragma unroll
        for (int i = 0; i < 8; ++i) sbuf[(rb + i) * 512 + col] = acc[i] * 0.0625f;
      }
    }
    __syncthreads();

    // --- 4. top-8 + softmax, one wave per batch row -----------------------
    {
      const int r = wid;
      float s[16];
#pragma unroll
      for (int i = 0; i < 16; ++i) s[i] = sbuf[r * 512 + lane * 16 + i];
      float tv[8];
      int   ti[8];
#pragma unroll
      for (int j = 0; j < 8; ++j) {
        float m = -1e30f;
        int   mi = 0;
#pragma unroll
        for (int i = 0; i < 16; ++i) {
          float v = s[i];
          if (v > m) { m = v; mi = lane * 16 + i; }
        }
#pragma unroll
        for (int off = 16; off > 0; off >>= 1) {
          float om = __shfl_xor(m, off);
          int   oi = __shfl_xor(mi, off);
          if (om > m || (om == m && oi < mi)) { m = om; mi = oi; }
        }
        tv[j] = m;
        ti[j] = mi;
        if ((mi >> 4) == lane) s[mi & 15] = -1e30f;  // knock out winner
      }
      if (lane == 0) {
        float mx = tv[0], sum = 0.f, e[8];
#pragma unroll
        for (int j = 0; j < 8; ++j) { e[j] = __expf(tv[j] - mx); sum += e[j]; }
        float inv = 1.f / sum;
#pragma unroll
        for (int j = 0; j < 8; ++j) { wbuf[r * 8 + j] = e[j] * inv; ibuf[r * 8 + j] = ti[j]; }
      }
    }
    __syncthreads();

    // --- 5. retrieved[r] = sum_j w_j * vals[idx_j] -> cbuf[r][512:1024] ---
    {
      const int r = wid;
      float w0[8];
      int   i0[8];
#pragma unroll
      for (int j = 0; j < 8; ++j) { w0[j] = wbuf[r * 8 + j]; i0[j] = ibuf[r * 8 + j]; }
#pragma clang loop unroll(disable)
      for (int c = 0; c < 16; ++c) {
        int d = c * 32 + lane;
        float acc = 0.f;
#pragma unroll
        for (int j = 0; j < 8; ++j) acc += w0[j] * vals[(size_t)i0[j] * 512 + d];
        cbuf[r * 1024 + 512 + d] = f2bf(acc);
      }
    }
    __syncthreads();

    // --- 6. h = tanh(concat @ Wh + bh)  (M=32,N=512,K=1024) ---------------
    {
#pragma clang loop unroll(disable)
      for (int rep = 0; rep < 2; ++rep) {
        const int tt = wid + rep * 32;
        const int mf = tt >> 5, nf = tt & 31;
        v8f acc = {};
#pragma clang loop unroll(disable)
        for (int kc = 0; kc < 32; ++kc) {
          v16bf a = load_a16x32(cbuf, mf * 16, kc * 32, 1024, lane);
          v16bf b = load_b32x16(WhT, nf * 16, kc * 32, 1024, lane);
          acc = wmma_bf16(a, b, acc);
        }
        const int col = nf * 16 + (lane & 15);
        const int rb  = mf * 16 + ((lane >> 4) << 3);
        const float bb = bh[col];
#pragma unroll
        for (int i = 0; i < 8; ++i) {
          float hv = tanhf(acc[i] + bb);
          sbuf[(rb + i) * 512 + col] = hv;                       // reuse sbuf as h
          hA[(size_t)(t * 32 + rb + i) * 512 + col] = f2bf(hv);  // for logits GEMM
        }
      }
    }
    __syncthreads();

    // --- 7. batch means of x and h ---------------------------------------
    if (tid < 512) {
      float s = 0.f;
#pragma clang loop unroll_count(4)
      for (int r = 0; r < 32; ++r) s += bf2f(cbuf[r * 1024 + tid]);
      meanx[tid] = s * (1.f / 32.f);
    } else {
      int e = tid - 512;
      float s = 0.f;
#pragma clang loop unroll_count(4)
      for (int r = 0; r < 32; ++r) s += sbuf[r * 512 + e];
      meanh[e] = s * (1.f / 32.f);
    }
    __syncthreads();

    // --- 8. cache write at slot ptr=t (N=512 > T=64, no wrap) -------------
    if (tid < 256) {
      float a = 0.f;
#pragma clang loop unroll_count(4)
      for (int e = 0; e < 512; ++e) a += meanx[e] * Wk[e * 256 + tid];
      keys_bf[t * 256 + tid] = f2bf(a);
    } else if (tid < 768) {
      const int j = tid - 256;
      float a = 0.f;
#pragma clang loop unroll_count(4)
      for (int e = 0; e < 512; ++e) a += meanh[e] * Wv[e * 512 + j];
      vals[(size_t)t * 512 + j] = a;
    }
    __threadfence();   // make global cache writes visible block-wide
    __syncthreads();
  }
}

// ---------------------------------------------------------------------------
// phase 2: logits[2048][32000] = hA[2048][512] @ Wout[512][32000] + bout
// block = 256 threads (8 waves), tile 128x128; wave tile 64x32 (4x2 frags)
// ---------------------------------------------------------------------------
__global__ __launch_bounds__(256) void logits_gemm(
    const unsigned short* __restrict__ hA,     // [2048][512] bf16
    const unsigned short* __restrict__ WoutT,  // [32000][512] bf16
    const float* __restrict__ bout, float* __restrict__ out) {
  const int lane = threadIdx.x & 31;
  const int wid  = threadIdx.x >> 5;
  const int m0 = blockIdx.y * 128 + (wid >> 2) * 64;
  const int n0 = blockIdx.x * 128 + (wid & 3) * 32;

  v8f acc[4][2] = {};
#pragma clang loop unroll_count(2)
  for (int kc = 0; kc < 16; ++kc) {
    const int k0 = kc * 32;
    // hint next B chunk into cache (global_prefetch_b8)
    __builtin_prefetch((const void*)(WoutT + (size_t)(n0 + lane) * 512 + k0 + 32), 0, 3);
    v16bf a[4], b[2];
#pragma unroll
    for (int i = 0; i < 4; ++i) a[i] = load_a16x32(hA, m0 + i * 16, k0, 512, lane);
#pragma unroll
    for (int j = 0; j < 2; ++j) b[j] = load_b32x16(WoutT, n0 + j * 16, k0, 512, lane);
#pragma unroll
    for (int i = 0; i < 4; ++i)
#pragma unroll
      for (int j = 0; j < 2; ++j) acc[i][j] = wmma_bf16(a[i], b[j], acc[i][j]);
  }

  const int colB = lane & 15;
  const int rsel = (lane >> 4) << 3;
#pragma unroll
  for (int j = 0; j < 2; ++j) {
    const int n  = n0 + j * 16 + colB;
    const float bb = bout[n];
#pragma unroll
    for (int i = 0; i < 4; ++i) {
#pragma unroll
      for (int r = 0; r < 8; ++r) {
        const int m = m0 + i * 16 + rsel + r;
        out[(size_t)m * 32000 + n] = acc[i][j][r] + bb;
      }
    }
  }
}

// ---------------------------------------------------------------------------
// launch
// ---------------------------------------------------------------------------
extern "C" void kernel_launch(void* const* d_in, const int* in_sizes, int n_in,
                              void* d_out, int out_size, void* d_ws, size_t ws_size,
                              hipStream_t stream) {
  const int*   tokens = (const int*)  d_in[0];
  const float* emb    = (const float*)d_in[1];
  const float* Wq     = (const float*)d_in[2];
  const float* Wk     = (const float*)d_in[3];
  const float* Wv     = (const float*)d_in[4];
  const float* Wh     = (const float*)d_in[5];
  const float* bh     = (const float*)d_in[6];
  const float* Wout   = (const float*)d_in[7];
  const float* bout   = (const float*)d_in[8];
  const float* keys0  = (const float*)d_in[9];
  const float* vals0  = (const float*)d_in[10];
  (void)in_sizes; (void)n_in; (void)out_size; (void)ws_size;

  char* ws = (char*)d_ws;
  size_t off = 0;
  unsigned short* WoutT = (unsigned short*)(ws + off); off += (size_t)32000 * 512 * 2;
  unsigned short* WqT   = (unsigned short*)(ws + off); off += (size_t)256 * 512 * 2;
  unsigned short* WhT   = (unsigned short*)(ws + off); off += (size_t)512 * 1024 * 2;
  unsigned short* keysb = (unsigned short*)(ws + off); off += (size_t)512 * 256 * 2;
  float*          vals  = (float*)         (ws + off); off += (size_t)512 * 512 * 4;
  unsigned short* hA    = (unsigned short*)(ws + off); off += (size_t)2048 * 512 * 2;

  const dim3 tb(32, 8);
  // Wout [512][32000] -> WoutT [32000][512]
  transpose_f32_to_bf16<<<dim3(32000 / 32, 512 / 32), tb, 0, stream>>>(Wout, WoutT, 512, 32000);
  // Wq [512][256] -> WqT [256][512]
  transpose_f32_to_bf16<<<dim3(256 / 32, 512 / 32), tb, 0, stream>>>(Wq, WqT, 512, 256);
  // Wh [1024][512] -> WhT [512][1024]
  transpose_f32_to_bf16<<<dim3(512 / 32, 1024 / 32), tb, 0, stream>>>(Wh, WhT, 1024, 512);
  // keys cache bf16 mirror; vals cache f32 working copy
  cvt_f32_to_bf16<<<(512 * 256 + 255) / 256, 256, 0, stream>>>(keys0, keysb, 512 * 256);
  hipMemcpyAsync(vals, vals0, (size_t)512 * 512 * 4, hipMemcpyDeviceToDevice, stream);

  cran_recurrent<<<1, 1024, 0, stream>>>(tokens, emb, WqT, WhT, bh, Wk, Wv,
                                         keysb, vals, hA);

  logits_gemm<<<dim3(32000 / 128, 2048 / 128), 256, 0, stream>>>(
      hA, WoutT, bout, (float*)d_out);
}
